// MambaRecursiveStateAdjustmentV5_41944650613244
// MI455X (gfx1250) — matrix operand order repaired and checked by reading, not compiled, not run on hardware
//
#include <hip/hip_runtime.h>
#include <cstddef>

// ---------------- problem dims ----------------
#define BB    2
#define TITER 3
#define CC    48
#define HH    48
#define WW    48
#define LL    (HH*WW)        // 2304
#define DIN   96
#define DST   8
#define KK    4
#define DTR   3
#define EPSF  1e-6f
#define NCHW  (BB*CC*LL)     // 221184
#define NT    (LL/16)        // 144 pixel tiles (each within one row: 48 = 3*16)
#define KPAD  576            // conv K = tap*64 + ci, 9 taps x 64 (48 used)
#define NKC   (KPAD/32)      // 18 WMMA K-chunks per conv

// ---------------- WMMA plumbing (CDNA5, wave32) ----------------
typedef __attribute__((ext_vector_type(16))) _Float16 v16h;
typedef __attribute__((ext_vector_type(8)))  _Float16 v8h;
typedef __attribute__((ext_vector_type(8)))  float    v8f;

__device__ __forceinline__ v8f wmma16(v16h a, v16h b, v8f c) {
  return __builtin_amdgcn_wmma_f32_16x16x32_f16(false, a, false, b, (short)0, c,
                                                false, false);
}
// A fragment: row-packed f16 weights wrow[k] (k = HW K index). Lane layout:
// element e<8 -> k = kc*32 + half*8 + e ; e>=8 -> k = kc*32 + 16 + half*8 + (e-8)
__device__ __forceinline__ v16h frag_a(const _Float16* wrow, int kc, int half) {
  v8h a0 = *(const v8h*)(wrow + kc * 32 + half * 8);
  v8h a1 = *(const v8h*)(wrow + kc * 32 + 16 + half * 8);
  return __builtin_shufflevector(a0, a1, 0,1,2,3,4,5,6,7,8,9,10,11,12,13,14,15);
}
// B fragment: per-pixel K-major row brow[k]; lane elements e<8 -> k = kc*32 + half*8 + e ; e>=8 -> k = kc*32 + 16 + half*8 + (e-8)
__device__ __forceinline__ v16h frag_b(const _Float16* brow, int kc, int half) {
  v8h b0 = *(const v8h*)(brow + kc * 32 + half * 8);
  v8h b1 = *(const v8h*)(brow + kc * 32 + 16 + half * 8);
  return __builtin_shufflevector(b0, b1, 0,1,2,3,4,5,6,7,8,9,10,11,12,13,14,15);
}
__device__ __forceinline__ float sigm(float x) { return 1.f / (1.f + expf(-x)); }

// ---------------- Tensor Data Mover (gfx1250) ----------------
#if defined(__has_builtin)
#if __has_builtin(__builtin_amdgcn_tensor_load_to_lds)
#define HAVE_TDM 1
#endif
#endif
#ifndef HAVE_TDM
#define HAVE_TDM 0
#endif

#if HAVE_TDM
typedef unsigned int v4u_ __attribute__((ext_vector_type(4)));
typedef int          v8i_ __attribute__((ext_vector_type(8)));
typedef int          v4i_ __attribute__((ext_vector_type(4)));

// DMA a 48(x) x 3(rows) x 48(ch) f32 slab (row stride 48, ch stride 2304) to LDS.
__device__ __forceinline__ void tdm_load_slab(const float* gsrc, unsigned lds_off) {
  unsigned long long ga = (unsigned long long)gsrc;
  v4u_ g0 = { 1u,                                   // count=1, user descriptor
              lds_off,                              // lds_addr (bytes)
              (unsigned)(ga & 0xffffffffu),         // global_addr[31:0]
              (unsigned)((ga >> 32) & 0x1ffffffu) | (2u << 30) };  // addr[56:32]|type=2
  v8i_ g1 = { (int)(2u << 16),                      // data_size = 4B
              (int)(48u << 16),                     // tensor_dim0 = 48 (lo16)
              (int)(48u << 16),                     // tensor_dim1 = 48 (lo16)
              (int)(48u << 16),                     // tile_dim0 = 48
              (int)(3u | (48u << 16)),              // tile_dim1 = 3, tile_dim2 = 48
              48,                                   // tensor_dim0_stride = 48
              (int)(2304u << 16),                   // tensor_dim1_stride = 2304 (lo16)
              0 };
  v4i_ g2 = { 48, 0, 0, 0 };                        // tensor_dim2 = 48
  v4i_ g3 = { 0, 0, 0, 0 };
#if __clang_major__ >= 23
  v8i_ g4 = { 0, 0, 0, 0, 0, 0, 0, 0 };
  __builtin_amdgcn_tensor_load_to_lds(g0, g1, g2, g3, g4, 0);
#else
  __builtin_amdgcn_tensor_load_to_lds(g0, g1, g2, g3, 0);
#endif
}
#endif

// ---------------- conv staging: slab (TDM) -> f16 im2col in LDS ----------------
// slab layout: [ci][row 0..2][x 0..47]; im layout: [pix 0..15][k 0..575]
__device__ __forceinline__ void stage_conv(const float* __restrict__ xb,
                                           int py, int pxb,
                                           float* slab, _Float16* im) {
  int row0 = py - 1;
#if HAVE_TDM
  if (py >= 1 && py <= 46) {          // interior: rows row0..row0+2 all in-bounds
    if (threadIdx.x < 32) {
      tdm_load_slab(xb + (size_t)row0 * WW,
                    (unsigned)(unsigned long long)slab);
      __builtin_amdgcn_s_wait_tensorcnt(0);
    }
  } else
#endif
  {
    for (int idx = threadIdx.x; idx < CC * 3 * WW; idx += 96) {
      int x0 = idx % WW, r = (idx / WW) % 3, ci = idx / (3 * WW);
      int yy = row0 + r;
      slab[idx] = ((unsigned)yy < HH) ? xb[(size_t)ci * LL + (size_t)yy * WW + x0] : 0.f;
    }
  }
  __syncthreads();
  for (int idx = threadIdx.x; idx < 16 * KPAD; idx += 96) {
    int k = idx >> 4, pix = idx & 15;
    int ci = k & 63, tap = k >> 6;          // k = tap*64 + ci (shift/AND decode)
    int ky = tap / 3, kx = tap - ky * 3;
    int xx = pxb + pix + kx - 1;
    float v = 0.f;
    if (ci < CC && (unsigned)xx < WW) v = slab[ci * (3 * WW) + ky * WW + xx];
    im[pix * KPAD + k] = (_Float16)v;
  }
  __syncthreads();
}

__device__ __forceinline__ v8f conv_mm(const _Float16* __restrict__ wp,
                                       const _Float16* im,
                                       int wv, int half, int ln) {
  const _Float16* wrow = wp + (size_t)(wv * 16 + ln) * KPAD;
  const _Float16* brow = im + ln * KPAD;
  v8f acc = {};
#pragma unroll
  for (int kc = 0; kc < NKC; ++kc)
    acc = wmma16(frag_a(wrow, kc, half), frag_b(brow, kc, half), acc);
  return acc;
}

// conv3x3 + bias -> LayerNorm2d(affine) -> act (0=sigmoid, 1=leaky0.01)
__global__ void __launch_bounds__(96)
k_conv_ln_act(const float* __restrict__ xin, const _Float16* __restrict__ wp,
              const float* __restrict__ bia, const float* __restrict__ gam,
              const float* __restrict__ bet, float* __restrict__ out, int act) {
  __shared__ __align__(16) float    slab[CC * 3 * WW];
  __shared__ __align__(16) _Float16 im[16 * KPAD];
  __shared__ float sh[CC][16];
  __shared__ float smu[16], srs[16];

  int tile = blockIdx.x % NT, b = blockIdx.x / NT;
  int wv = threadIdx.x >> 5, lane = threadIdx.x & 31;
  int half = lane >> 4, ln = lane & 15;
  int pixb = tile * 16, py = pixb / WW, pxb = pixb % WW;

  stage_conv(xin + (size_t)b * CC * LL, py, pxb, slab, im);
  v8f acc = conv_mm(wp, im, wv, half, ln);

#pragma unroll
  for (int r = 0; r < 8; ++r) {
    int co = wv * 16 + half * 8 + r;
    sh[co][ln] = acc[r] + bia[co];
  }
  __syncthreads();
  if (threadIdx.x < 16) {
    float s = 0.f, s2 = 0.f;
    for (int c = 0; c < CC; ++c) { float v = sh[c][threadIdx.x]; s += v; s2 += v * v; }
    float m = s * (1.f / CC);
    smu[threadIdx.x] = m;
    srs[threadIdx.x] = rsqrtf(s2 * (1.f / CC) - m * m + EPSF);
  }
  __syncthreads();
  float* ob = out + (size_t)b * CC * LL;
  for (int idx = threadIdx.x; idx < CC * 16; idx += 96) {
    int c = idx >> 4, pc = idx & 15;
    float v = (sh[c][pc] - smu[pc]) * srs[pc] * gam[c] + bet[c];
    v = (act == 0) ? 1.f / (1.f + expf(-v)) : (v > 0.f ? v : 0.01f * v);
    ob[(size_t)c * LL + pixb + pc] = v;
  }
}

// conv3x3 + bias + residual (post: 1 = sigmoid)
__global__ void __launch_bounds__(96)
k_conv_res(const float* __restrict__ hin, const _Float16* __restrict__ wp,
           const float* __restrict__ bia, const float* __restrict__ xres,
           float* __restrict__ out, int post) {
  __shared__ __align__(16) float    slab[CC * 3 * WW];
  __shared__ __align__(16) _Float16 im[16 * KPAD];

  int tile = blockIdx.x % NT, b = blockIdx.x / NT;
  int wv = threadIdx.x >> 5, lane = threadIdx.x & 31;
  int half = lane >> 4, ln = lane & 15;
  int pixb = tile * 16, py = pixb / WW, pxb = pixb % WW;

  stage_conv(hin + (size_t)b * CC * LL, py, pxb, slab, im);
  v8f acc = conv_mm(wp, im, wv, half, ln);

  const float* rb = xres + (size_t)b * CC * LL;
  float* ob = out + (size_t)b * CC * LL;
#pragma unroll
  for (int r = 0; r < 8; ++r) {
    int co = wv * 16 + half * 8 + r;
    float v = acc[r] + bia[co] + rb[(size_t)co * LL + pixb + ln];
    if (post) v = 1.f / (1.f + expf(-v));
    ob[(size_t)co * LL + pixb + ln] = v;
  }
}

// ---------------- weight pack kernels (f32 -> f16 fragment layouts) ----------------
__global__ void k_pack_conv(const float* __restrict__ w, _Float16* __restrict__ wp) {
  int i = blockIdx.x * 256 + threadIdx.x;
  if (i >= CC * KPAD) return;
  int k = i % KPAD, co = i / KPAD;
  int ci = k & 63, tap = k >> 6;
  wp[i] = (_Float16)((ci < CC) ? w[((size_t)co * CC + ci) * 9 + tap] : 0.f);
}
__global__ void k_pack_inw(const float* __restrict__ w, _Float16* __restrict__ wp) {
  int i = blockIdx.x * 256 + threadIdx.x;       // 192 x 64 (K padded 48->64)
  if (i >= 2 * DIN * 64) return;
  int k = i & 63, m = i >> 6;
  wp[i] = (_Float16)((k < CC) ? w[m * CC + k] : 0.f);
}
__global__ void k_pack_xpw(const float* __restrict__ w, _Float16* __restrict__ wp) {
  int i = blockIdx.x * 256 + threadIdx.x;       // 4 x 32 x 96 (M padded 19->32)
  if (i >= KK * 32 * DIN) return;
  int k = i % DIN, r = (i / DIN) % 32, kd = i / (32 * DIN);
  wp[i] = (_Float16)((r < 19) ? w[((size_t)kd * 19 + r) * DIN + k] : 0.f);
}
__global__ void k_pack_f16(const float* __restrict__ w, _Float16* __restrict__ wp, int n) {
  int i = blockIdx.x * 256 + threadIdx.x;
  if (i < n) wp[i] = (_Float16)w[i];
}

// ---------------- LayerNorm2d / elementwise ----------------
__global__ void k_ln2d(const float* __restrict__ x, size_t bstride,
                       const float* __restrict__ gam, const float* __restrict__ bet,
                       float* __restrict__ out) {
  int i = blockIdx.x * blockDim.x + threadIdx.x;
  if (i >= BB * LL) return;
  int b = i / LL, p = i % LL;
  const float* xb = x + (size_t)b * bstride;
  float s = 0.f, s2 = 0.f;
  for (int c = 0; c < CC; ++c) { float v = xb[(size_t)c * LL + p]; s += v; s2 += v * v; }
  float m = s * (1.f / CC), rs = rsqrtf(s2 * (1.f / CC) - m * m + EPSF);
  float* ob = out + (size_t)b * CC * LL;
  for (int c = 0; c < CC; ++c) {
    float v = (xb[(size_t)c * LL + p] - m) * rs;
    if (gam) v = v * gam[c] + bet[c];
    ob[(size_t)c * LL + p] = v;
  }
}

__global__ void k_prev_init(const float* __restrict__ t, const float* __restrict__ gam,
                            const float* __restrict__ bet, const float* __restrict__ cur,
                            float* __restrict__ prev) {
  int i = blockIdx.x * blockDim.x + threadIdx.x;
  if (i >= BB * LL) return;
  int b = i / LL, p = i % LL;
  const float* tb = t + (size_t)b * CC * LL;
  float s = 0.f, s2 = 0.f;
  for (int c = 0; c < CC; ++c) { float v = -tb[(size_t)c * LL + p]; s += v; s2 += v * v; }
  float m = s * (1.f / CC), rs = rsqrtf(s2 * (1.f / CC) - m * m + EPSF);
  const float* cb = cur + (size_t)b * CC * LL;
  float* pb = prev + (size_t)b * CC * LL;
  for (int c = 0; c < CC; ++c) {
    float v = (-tb[(size_t)c * LL + p] - m) * rs * gam[c] + bet[c];
    pb[(size_t)c * LL + p] = cb[(size_t)c * LL + p] * expf(-v);
  }
}

__global__ void k_kl(const float* __restrict__ a, const float* __restrict__ b,
                     float* __restrict__ o, int n) {
  int i = blockIdx.x * blockDim.x + threadIdx.x;
  if (i < n) o[i] = a[i] * (logf(a[i] + EPSF) - logf(b[i] + EPSF));
}
__global__ void k_update(const float* __restrict__ cur, const float* __restrict__ ss,
                         const float* __restrict__ si, const float* __restrict__ bi,
                         float* __restrict__ o, int n) {
  int i = blockIdx.x * blockDim.x + threadIdx.x;
  if (i < n) o[i] = cur[i] * expf(ss[i] + si[i]) + bi[i];
}

// ---------------- SS2D kernels ----------------
// xz(b,192,L) = in_w(192x48) @ x(b,48,L) + in_b  — one block per 16-pixel tile
__global__ void __launch_bounds__(128)
k_inproj(const float* __restrict__ x, const _Float16* __restrict__ wp,
         const float* __restrict__ bia, float* __restrict__ xz) {
  __shared__ __align__(16) _Float16 im[16 * 64];
  int nt = blockIdx.x % NT, b = blockIdx.x / NT;
  int pixb = nt * 16;
  for (int idx = threadIdx.x; idx < 16 * 64; idx += 128) {
    int k = idx >> 4, pix = idx & 15;
    im[pix * 64 + k] = (_Float16)((k < CC) ? x[((size_t)b * CC + k) * LL + pixb + pix] : 0.f);
  }
  __syncthreads();
  int wv = threadIdx.x >> 5, lane = threadIdx.x & 31;
  int half = lane >> 4, ln = lane & 15;
  const _Float16* brow = im + ln * 64;
  for (int mg = 0; mg < 3; ++mg) {
    int mt = mg * 4 + wv;
    const _Float16* wrow = wp + (size_t)(mt * 16 + ln) * 64;
    v8f acc = {};
#pragma unroll
    for (int kc = 0; kc < 2; ++kc)
      acc = wmma16(frag_a(wrow, kc, half), frag_b(brow, kc, half), acc);
#pragma unroll
    for (int r = 0; r < 8; ++r) {
      int m = mt * 16 + half * 8 + r;
      xz[((size_t)b * 2 * DIN + m) * LL + pixb + ln] = acc[r] + bia[m];
    }
  }
}

// depthwise 3x3 + SiLU on xi = xz[:, :96]
__global__ void k_dwsilu(const float* __restrict__ xz, const float* __restrict__ cw,
                         const float* __restrict__ cb, float* __restrict__ xi) {
  int i = blockIdx.x * blockDim.x + threadIdx.x;
  if (i >= BB * DIN * LL) return;
  int p = i % LL, d = (i / LL) % DIN, b = i / (LL * DIN);
  int y = p / WW, x0 = p % WW;
  const float* src = xz + ((size_t)b * 2 * DIN + d) * LL;
  float s = cb[d];
  for (int ky = 0; ky < 3; ++ky)
    for (int kx = 0; kx < 3; ++kx) {
      int yy = y + ky - 1, xx = x0 + kx - 1;
      if ((unsigned)yy < HH && (unsigned)xx < WW)
        s += src[yy * WW + xx] * cw[d * 9 + ky * 3 + kx];
    }
  xi[((size_t)b * DIN + d) * LL + p] = s * sigm(s);
}

// xs(b,4,96,L): 0=row scan, 1=col scan, 2=flip(0), 3=flip(1)
__global__ void k_build_xs(const float* __restrict__ xi, float* __restrict__ xs) {
  int i = blockIdx.x * blockDim.x + threadIdx.x;
  if (i >= BB * DIN * LL) return;
  int p = i % LL, d = (i / LL) % DIN, b = i / (LL * DIN);
  int lt = (p % WW) * HH + p / WW;
  float v0 = xi[((size_t)b * DIN + d) * LL + p];
  float v1 = xi[((size_t)b * DIN + d) * LL + lt];
  size_t base = ((size_t)b * KK) * DIN * LL + (size_t)d * LL;
  xs[base + 0 * (size_t)DIN * LL + p]            = v0;
  xs[base + 1 * (size_t)DIN * LL + p]            = v1;
  xs[base + 2 * (size_t)DIN * LL + (LL - 1 - p)] = v0;
  xs[base + 3 * (size_t)DIN * LL + (LL - 1 - p)] = v1;
}

// dbl(b,k,19,L) = xp_w[k](19x96, M-padded) @ xs[b,k]
__global__ void __launch_bounds__(64)
k_xproj(const float* __restrict__ xs, const _Float16* __restrict__ wpx,
        float* __restrict__ dbl) {
  __shared__ __align__(16) _Float16 im[16 * DIN];
  int g = blockIdx.x;
  int nt = g % NT; g /= NT;
  int kd = g % KK, b = g / KK;
  int pixb = nt * 16;
  const float* xsb = xs + ((size_t)b * KK + kd) * DIN * LL;
  for (int idx = threadIdx.x; idx < 16 * DIN; idx += 64) {
    int k = idx >> 4, pix = idx & 15;
    im[pix * DIN + k] = (_Float16)xsb[(size_t)k * LL + pixb + pix];
  }
  __syncthreads();
  int wv = threadIdx.x >> 5, lane = threadIdx.x & 31;
  int half = lane >> 4, ln = lane & 15;
  const _Float16* wrow = wpx + ((size_t)kd * 32 + wv * 16 + ln) * DIN;
  const _Float16* brow = im + ln * DIN;
  v8f acc = {};
#pragma unroll
  for (int kc = 0; kc < 3; ++kc)
    acc = wmma16(frag_a(wrow, kc, half), frag_b(brow, kc, half), acc);
#pragma unroll
  for (int r = 0; r < 8; ++r) {
    int m = wv * 16 + half * 8 + r;
    if (m < 19) dbl[(((size_t)b * KK + kd) * 19 + m) * LL + pixb + ln] = acc[r];
  }
}

// dt(b,k,96,L) = softplus(dt_w[k](96x3) @ dbl[:,:,:3] + dt_b[k])
__global__ void k_dt(const float* __restrict__ dbl, const float* __restrict__ dtw,
                     const float* __restrict__ dtb, float* __restrict__ dt) {
  int i = blockIdx.x * blockDim.x + threadIdx.x;
  if (i >= BB * KK * DIN * LL) return;
  int l = i % LL, d = (i / LL) % DIN, kd = (i / (LL * DIN)) % KK, b = i / (LL * DIN * KK);
  float s = dtb[kd * DIN + d];
  const float* db = dbl + (((size_t)b * KK + kd) * 19) * LL;
  for (int r = 0; r < DTR; ++r)
    s += dtw[((size_t)kd * DIN + d) * DTR + r] * db[(size_t)r * LL + l];
  dt[i] = s > 20.f ? s : log1pf(expf(s));
}

// selective scan: block per (b,k); thread per d; B/C staged via LDS in 64-step chunks
__global__ void __launch_bounds__(96)
k_scan(const float* __restrict__ dt, const float* __restrict__ xs,
       const float* __restrict__ dbl, const float* __restrict__ Alog,
       const float* __restrict__ Dsk, float* __restrict__ ys) {
  int b = blockIdx.x / KK, kd = blockIdx.x % KK;
  int d = threadIdx.x;
  float Ak[DST], h[DST];
#pragma unroll
  for (int n = 0; n < DST; ++n) {
    Ak[n] = -expf(Alog[((size_t)kd * DIN + d) * DST + n]);
    h[n] = 0.f;
  }
  float dsv = Dsk[kd * DIN + d];
  const float* dtp = dt  + (((size_t)b * KK + kd) * DIN + d) * LL;
  const float* up  = xs  + (((size_t)b * KK + kd) * DIN + d) * LL;
  const float* Bp  = dbl + (((size_t)b * KK + kd) * 19 + DTR) * LL;
  const float* Cp  = dbl + (((size_t)b * KK + kd) * 19 + DTR + DST) * LL;
  float* yp        = ys  + (((size_t)b * KK + kd) * DIN + d) * LL;

  __shared__ float sB[64][DST], sC[64][DST];
  for (int l0 = 0; l0 < LL; l0 += 64) {
    __syncthreads();
    for (int idx = threadIdx.x; idx < 64 * DST; idx += 96) {
      int dl = idx >> 3, n = idx & 7;
      sB[dl][n] = Bp[(size_t)n * LL + l0 + dl];
      sC[dl][n] = Cp[(size_t)n * LL + l0 + dl];
    }
    if (l0 + 64 < LL && threadIdx.x < DST) {   // prefetch next B/C chunk
      __builtin_prefetch(Bp + (size_t)threadIdx.x * LL + l0 + 64, 0, 0);
      __builtin_prefetch(Cp + (size_t)threadIdx.x * LL + l0 + 64, 0, 0);
    }
    __syncthreads();
    for (int dl = 0; dl < 64; ++dl) {
      int l = l0 + dl;
      float dtv = dtp[l], uv = up[l];
      float du = dtv * uv, yv = 0.f;
#pragma unroll
      for (int n = 0; n < DST; ++n) {
        h[n] = h[n] * expf(dtv * Ak[n]) + du * sB[dl][n];
        yv += h[n] * sC[dl][n];
      }
      yp[l] = yv + uv * dsv;
    }
  }
}

// combine 4 scan directions -> y(b,96,L)
__global__ void k_combine(const float* __restrict__ ys, float* __restrict__ y) {
  int i = blockIdx.x * blockDim.x + threadIdx.x;
  if (i >= BB * DIN * LL) return;
  int p = i % LL, d = (i / LL) % DIN, b = i / (LL * DIN);
  int lt = (p % WW) * HH + p / WW;
  size_t base = ((size_t)b * KK) * DIN * LL + (size_t)d * LL;
  float v = ys[base + p]
          + ys[base + 2 * (size_t)DIN * LL + (LL - 1 - p)]
          + ys[base + 1 * (size_t)DIN * LL + lt]
          + ys[base + 3 * (size_t)DIN * LL + (LL - 1 - lt)];
  y[((size_t)b * DIN + d) * LL + p] = v;
}

// LN over 96 channels + SiLU(z) gate, in place on y
__global__ void k_gate_ln(float* __restrict__ y, const float* __restrict__ xz,
                          const float* __restrict__ g, const float* __restrict__ be) {
  int i = blockIdx.x * blockDim.x + threadIdx.x;
  if (i >= BB * LL) return;
  int b = i / LL, p = i % LL;
  float s = 0.f, s2 = 0.f;
  for (int d = 0; d < DIN; ++d) {
    float v = y[((size_t)b * DIN + d) * LL + p];
    s += v; s2 += v * v;
  }
  float m = s * (1.f / DIN), rs = rsqrtf(s2 * (1.f / DIN) - m * m + EPSF);
  for (int d = 0; d < DIN; ++d) {
    float zn = xz[((size_t)b * 2 * DIN + DIN + d) * LL + p];
    float v = (y[((size_t)b * DIN + d) * LL + p] - m) * rs * g[d] + be[d];
    y[((size_t)b * DIN + d) * LL + p] = v * zn * sigm(zn);
  }
}

// out(b,48,L) = out_w(48x96) @ y + out_b
__global__ void __launch_bounds__(96)
k_outproj(const float* __restrict__ y, const _Float16* __restrict__ wp,
          const float* __restrict__ bia, float* __restrict__ out) {
  __shared__ __align__(16) _Float16 im[16 * DIN];
  int nt = blockIdx.x % NT, b = blockIdx.x / NT;
  int pixb = nt * 16;
  for (int idx = threadIdx.x; idx < 16 * DIN; idx += 96) {
    int k = idx >> 4, pix = idx & 15;
    im[pix * DIN + k] = (_Float16)y[((size_t)b * DIN + k) * LL + pixb + pix];
  }
  __syncthreads();
  int wv = threadIdx.x >> 5, lane = threadIdx.x & 31;
  int half = lane >> 4, ln = lane & 15;
  const _Float16* wrow = wp + (size_t)(wv * 16 + ln) * DIN;
  const _Float16* brow = im + ln * DIN;
  v8f acc = {};
#pragma unroll
  for (int kc = 0; kc < 3; ++kc)
    acc = wmma16(frag_a(wrow, kc, half), frag_b(brow, kc, half), acc);
#pragma unroll
  for (int r = 0; r < 8; ++r) {
    int co = wv * 16 + half * 8 + r;
    out[((size_t)b * CC + co) * LL + pixb + ln] = acc[r] + bia[co];
  }
}

// ---------------- host orchestration ----------------
struct SSP {
  const float *in_w, *in_b, *cw, *cb, *xp_w, *dt_w, *dt_b, *A_log, *Ds, *on_g, *on_b, *out_w, *out_b;
};
struct SSPack { _Float16 *inw, *xpw, *outw; };

static SSP get_ssp(void* const* d_in, int base) {
  SSP p;
  p.in_w  = (const float*)d_in[base + 0];  p.in_b  = (const float*)d_in[base + 1];
  p.cw    = (const float*)d_in[base + 2];  p.cb    = (const float*)d_in[base + 3];
  p.xp_w  = (const float*)d_in[base + 4];  p.dt_w  = (const float*)d_in[base + 5];
  p.dt_b  = (const float*)d_in[base + 6];  p.A_log = (const float*)d_in[base + 7];
  p.Ds    = (const float*)d_in[base + 8];  p.on_g  = (const float*)d_in[base + 9];
  p.on_b  = (const float*)d_in[base + 10]; p.out_w = (const float*)d_in[base + 11];
  p.out_b = (const float*)d_in[base + 12];
  return p;
}

static void run_ss2d(const float* x, const SSP& P, const SSPack& W, float* out,
                     float* xzb, float* xib, float* xsb, float* dblb,
                     float* dtb, float* ysb, float* yb, hipStream_t s) {
  const int n1 = BB * DIN * LL, n2 = BB * KK * DIN * LL;
  k_inproj  <<<BB * NT, 128, 0, s>>>(x, W.inw, P.in_b, xzb);
  k_dwsilu  <<<(n1 + 255) / 256, 256, 0, s>>>(xzb, P.cw, P.cb, xib);
  k_build_xs<<<(n1 + 255) / 256, 256, 0, s>>>(xib, xsb);
  k_xproj   <<<BB * KK * NT, 64, 0, s>>>(xsb, W.xpw, dblb);
  k_dt      <<<(n2 + 255) / 256, 256, 0, s>>>(dblb, P.dt_w, P.dt_b, dtb);
  k_scan    <<<BB * KK, 96, 0, s>>>(dtb, xsb, dblb, P.A_log, P.Ds, ysb);
  k_combine <<<(n1 + 255) / 256, 256, 0, s>>>(ysb, yb);
  k_gate_ln <<<(BB * LL + 255) / 256, 256, 0, s>>>(yb, xzb, P.on_g, P.on_b);
  k_outproj <<<BB * NT, 96, 0, s>>>(yb, W.outw, P.out_b, out);
}

extern "C" void kernel_launch(void* const* d_in, const int* in_sizes, int n_in,
                              void* d_out, int out_size, void* d_ws, size_t ws_size,
                              hipStream_t stream) {
  (void)in_sizes; (void)n_in; (void)out_size; (void)ws_size;
  const float* img_seq = (const float*)d_in[0];
  const float* dzone   = (const float*)d_in[1];
  const float* sigma   = (const float*)d_in[2];
  auto F = [&](int i) { return (const float*)d_in[i]; };

  char* base = (char*)d_ws;
  size_t off = 0;
  auto allocf = [&](size_t n) {
    off = (off + 15) & ~(size_t)15;
    float* p = (float*)(base + off); off += n * 4; return p;
  };
  auto alloch = [&](size_t n) {
    off = (off + 15) & ~(size_t)15;
    _Float16* p = (_Float16*)(base + off); off += n * 2; return p;
  };

  float* cur   = allocf(NCHW);
  float* prv   = allocf(NCHW);
  float* nxt   = allocf(NCHW);
  float* tA    = allocf(NCHW);
  float* tB    = allocf(NCHW);
  float* simg  = allocf(NCHW);
  float* sst   = allocf(NCHW);
  float* sbia  = allocf(NCHW);
  float* imgb  = allocf(NCHW);
  float* featb = allocf(NCHW);
  float* klb   = allocf(NCHW);
  float* xzb   = allocf((size_t)BB * 2 * DIN * LL);
  float* xib   = allocf((size_t)BB * DIN * LL);
  float* xsb   = allocf((size_t)BB * KK * DIN * LL);
  float* dblb  = allocf((size_t)BB * KK * 19 * LL);
  float* dtbuf = allocf((size_t)BB * KK * DIN * LL);
  float* ysb   = allocf((size_t)BB * KK * DIN * LL);
  float* yb    = allocf((size_t)BB * DIN * LL);

  // ---- pack weights to f16 fragment layouts ----
  const int convIdx[12] = {3, 7, 9, 13, 15, 19, 29, 33, 35, 39, 41, 45};
  _Float16* wpc[12];
  for (int i = 0; i < 12; ++i) {
    wpc[i] = alloch((size_t)CC * KPAD);
    k_pack_conv<<<(CC * KPAD + 255) / 256, 256, 0, stream>>>(F(convIdx[i]), wpc[i]);
  }
  const int sspBase[3] = {47, 60, 73};
  SSP    ssp[3];
  SSPack ssw[3];
  for (int j = 0; j < 3; ++j) {
    ssp[j] = get_ssp(d_in, sspBase[j]);
    ssw[j].inw  = alloch((size_t)2 * DIN * 64);
    ssw[j].xpw  = alloch((size_t)KK * 32 * DIN);
    ssw[j].outw = alloch((size_t)CC * DIN);
    k_pack_inw<<<(2 * DIN * 64 + 255) / 256, 256, 0, stream>>>(ssp[j].in_w, ssw[j].inw);
    k_pack_xpw<<<(KK * 32 * DIN + 255) / 256, 256, 0, stream>>>(ssp[j].xp_w, ssw[j].xpw);
    k_pack_f16<<<(CC * DIN + 255) / 256, 256, 0, stream>>>(ssp[j].out_w, ssw[j].outw, CC * DIN);
  }

  const int NP = (BB * LL + 255) / 256;
  const int NE = (NCHW + 255) / 256;

  // current = sigmoid(crb(dz, difficult_zone, sigmoid))
  k_conv_ln_act<<<BB * NT, 96, 0, stream>>>(dzone, wpc[0], F(4), F(5), F(6), tA, 0);
  k_conv_res  <<<BB * NT, 96, 0, stream>>>(tA, wpc[1], F(8), dzone, cur, 1);

  // t = crb(init, sigma, sigmoid); previous = current / exp(LN2d(-t, norm_init))
  k_conv_ln_act<<<BB * NT, 96, 0, stream>>>(sigma, wpc[2], F(10), F(11), F(12), tA, 0);
  k_conv_res  <<<BB * NT, 96, 0, stream>>>(tA, wpc[3], F(14), sigma, tB, 0);
  k_prev_init <<<NP, 256, 0, stream>>>(tB, F(21), F(22), cur, prv);

  for (int i = 0; i < TITER; ++i) {
    const float* xslice = img_seq + (size_t)i * CC * LL;  // batch stride = ITER*C*L
    k_ln2d<<<NP, 256, 0, stream>>>(xslice, (size_t)TITER * CC * LL,
                                   F(23 + 2 * i), F(24 + 2 * i), imgb);
    int cb = 29 + 6 * i;
    k_conv_ln_act<<<BB * NT, 96, 0, stream>>>(imgb, wpc[6 + 2 * i], F(cb + 1),
                                              F(cb + 2), F(cb + 3), tA, 1);
    k_conv_res  <<<BB * NT, 96, 0, stream>>>(tA, wpc[7 + 2 * i], F(cb + 5), imgb, tB, 0);
    k_ln2d<<<NP, 256, 0, stream>>>(tB, (size_t)CC * LL, nullptr, nullptr, featb);

    run_ss2d(featb, ssp[0], ssw[0], simg, xzb, xib, xsb, dblb, dtbuf, ysb, yb, stream);
    k_kl<<<NE, 256, 0, stream>>>(cur, prv, klb, NCHW);
    run_ss2d(klb,  ssp[1], ssw[1], sst,  xzb, xib, xsb, dblb, dtbuf, ysb, yb, stream);
    run_ss2d(cur,  ssp[2], ssw[2], sbia, xzb, xib, xsb, dblb, dtbuf, ysb, yb, stream);
    k_update<<<NE, 256, 0, stream>>>(cur, sst, simg, sbia, nxt, NCHW);
    float* t = prv; prv = cur; cur = nxt; nxt = t;
  }

  // out = crb(unc, current, sigmoid)
  k_conv_ln_act<<<BB * NT, 96, 0, stream>>>(cur, wpc[4], F(16), F(17), F(18), tA, 0);
  k_conv_res  <<<BB * NT, 96, 0, stream>>>(tA, wpc[5], F(20), cur, (float*)d_out, 0);
}
